// LSTMEncoder_44848048505211
// MI455X (gfx1250) — matrix operand order repaired and checked by reading, not compiled, not run on hardware
//
#include <hip/hip_runtime.h>
#include <math.h>

typedef __attribute__((ext_vector_type(16))) _Float16 v16h;
typedef __attribute__((ext_vector_type(8)))  _Float16 v8h;
typedef __attribute__((ext_vector_type(8)))  float    v8f;

union V16H { v16h v; v8h p[2]; _Float16 h[16]; };
union V8F  { v8f  v; float f[8]; };

#define HID 32
#define TLEN 512
#define DIN 6

#if __has_builtin(__builtin_amdgcn_tanhf)
__device__ __forceinline__ float tanh_f(float x) { return __builtin_amdgcn_tanhf(x); }
#else
// branch-free: tanh(x) = (1 - e^{-2x}) / (1 + e^{-2x})
__device__ __forceinline__ float tanh_f(float x) {
    float e = __expf(-2.0f * x);
    return (1.0f - e) * __builtin_amdgcn_rcpf(1.0f + e);
}
#endif
__device__ __forceinline__ float sigm_f(float x) { return 0.5f * tanh_f(0.5f * x) + 0.5f; }

__device__ __forceinline__ v8f splat8(float s) {
    v8f r = { s, s, s, s, s, s, s, s };
    return r;
}

// Per-lane A-matrix (16x32 f16) pack: two contiguous 8xf16 runs per lane:
//   halves 0..7  <- row[hi*8 + 0 .. 7]
//   halves 8..15 <- row[hi*8 + 16 .. 23]
__device__ __forceinline__ v16h packA(const _Float16* __restrict__ row, int hi) {
    V16H a;
    a.p[0] = *(const v8h*)(row + hi * 8);
    a.p[1] = *(const v8h*)(row + hi * 8 + 16);
    return a.v;
}

__global__ __launch_bounds__(32) void lstm2_ln_kernel(
    const float* __restrict__ x,
    const float* __restrict__ Wih0, const float* __restrict__ Whh0,
    const float* __restrict__ bih0, const float* __restrict__ bhh0,
    const float* __restrict__ Wih1, const float* __restrict__ Whh1,
    const float* __restrict__ bih1, const float* __restrict__ bhh1,
    const float* __restrict__ lng,  const float* __restrict__ lnb,
    float* __restrict__ out)
{
    const int lane = threadIdx.x & 31;
    const int l    = lane & 15;
    const int hi   = lane >> 4;
    const int b0   = blockIdx.x * 16;   // 16 batch rows per (single-wave) workgroup

    // f16 state buffers in A-friendly layout; f32 copy of final h2 for LayerNorm
    __shared__ alignas(16) _Float16 h1h[16][HID];
    __shared__ alignas(16) _Float16 h2h[16][HID];
    __shared__ alignas(16) _Float16 xbh[16][HID];  // cols >= DIN stay zero
    __shared__ alignas(16) float    h2f[16][HID];

    // ---- Weights resident in registers, B-matrix layout (lane = N column, halves = K).
    V16H Bih0[8], Bhh0[8], Bih1[8], Bhh1[8];
#pragma unroll
    for (int c = 0; c < 8; ++c) {
#pragma unroll
        for (int e = 0; e < 16; ++e) {
            int n = 16 * c + l;
            int k = (e < 8) ? (8 * hi + e) : (16 + 8 * hi + (e - 8));
            Bhh0[c].h[e] = (_Float16)Whh0[n * HID + k];
            Bih1[c].h[e] = (_Float16)Wih1[n * HID + k];
            Bhh1[c].h[e] = (_Float16)Whh1[n * HID + k];
            Bih0[c].h[e] = (k < DIN) ? (_Float16)Wih0[n * DIN + k] : (_Float16)0.0f;
        }
    }
    float bias0[8], bias1[8];
#pragma unroll
    for (int c = 0; c < 8; ++c) {
        int n = 16 * c + l;
        bias0[c] = bih0[n] + bhh0[n];
        bias1[c] = bih1[n] + bhh1[n];
    }

    // ---- Cell state in C/D layout: cs[half].f[r] <-> (m = r + 8*hi, j = 16*half + l)
    V8F cs1[2], cs2[2];
#pragma unroll
    for (int hf = 0; hf < 2; ++hf)
#pragma unroll
        for (int r = 0; r < 8; ++r) { cs1[hf].f[r] = 0.0f; cs2[hf].f[r] = 0.0f; }

#pragma unroll
    for (int e = 0; e < 16; ++e) {
        h1h[l][16 * hi + e] = (_Float16)0.0f;
        h2h[l][16 * hi + e] = (_Float16)0.0f;
        xbh[l][16 * hi + e] = (_Float16)0.0f;
    }
    __syncthreads();  // single-wave WG: barrier is S_NOP, acts as LDS fence

#pragma unroll 1
    for (int t = 0; t < TLEN; ++t) {
        // ---- stage x tile (16 rows x 6) into zero-padded f16 LDS tile
        const float* xp = x + ((size_t)(b0 + l) * TLEN + t) * DIN;
        if (lane < 16) {
#pragma unroll
            for (int d = 0; d < DIN; ++d) xbh[l][d] = (_Float16)xp[d];
            if (t + 1 < TLEN) __builtin_prefetch(xp + DIN, 0, 0);
        }
        __syncthreads();

        v16h Ax = packA(&xbh[l][0], hi);
        v16h A1 = packA(&h1h[l][0], hi);

        // ================= Layer 1 =================
#pragma unroll
        for (int hf = 0; hf < 2; ++hf) {
            v8f ai = __builtin_amdgcn_wmma_f32_16x16x32_f16(false, Ax, false, Bih0[hf    ].v, (short)0, splat8(bias0[hf    ]), false, false);
            ai     = __builtin_amdgcn_wmma_f32_16x16x32_f16(false, A1, false, Bhh0[hf    ].v, (short)0, ai, false, false);
            v8f af = __builtin_amdgcn_wmma_f32_16x16x32_f16(false, Ax, false, Bih0[hf + 2].v, (short)0, splat8(bias0[hf + 2]), false, false);
            af     = __builtin_amdgcn_wmma_f32_16x16x32_f16(false, A1, false, Bhh0[hf + 2].v, (short)0, af, false, false);
            v8f ag = __builtin_amdgcn_wmma_f32_16x16x32_f16(false, Ax, false, Bih0[hf + 4].v, (short)0, splat8(bias0[hf + 4]), false, false);
            ag     = __builtin_amdgcn_wmma_f32_16x16x32_f16(false, A1, false, Bhh0[hf + 4].v, (short)0, ag, false, false);
            v8f ao = __builtin_amdgcn_wmma_f32_16x16x32_f16(false, Ax, false, Bih0[hf + 6].v, (short)0, splat8(bias0[hf + 6]), false, false);
            ao     = __builtin_amdgcn_wmma_f32_16x16x32_f16(false, A1, false, Bhh0[hf + 6].v, (short)0, ao, false, false);
            V8F vi, vf, vg, vo; vi.v = ai; vf.v = af; vg.v = ag; vo.v = ao;
#pragma unroll
            for (int r = 0; r < 8; ++r) {
                float ig = sigm_f(vi.f[r]);
                float fg = sigm_f(vf.f[r]);
                float gg = tanh_f(vg.f[r]);
                float og = sigm_f(vo.f[r]);
                float cc = fg * cs1[hf].f[r] + ig * gg;
                cs1[hf].f[r] = cc;
                h1h[r + 8 * hi][16 * hf + l] = (_Float16)(og * tanh_f(cc));
            }
        }
        __syncthreads();

        v16h Ah1 = packA(&h1h[l][0], hi);
        v16h A2  = packA(&h2h[l][0], hi);

        // ================= Layer 2 =================
#pragma unroll
        for (int hf = 0; hf < 2; ++hf) {
            v8f ai = __builtin_amdgcn_wmma_f32_16x16x32_f16(false, Ah1, false, Bih1[hf    ].v, (short)0, splat8(bias1[hf    ]), false, false);
            ai     = __builtin_amdgcn_wmma_f32_16x16x32_f16(false, A2,  false, Bhh1[hf    ].v, (short)0, ai, false, false);
            v8f af = __builtin_amdgcn_wmma_f32_16x16x32_f16(false, Ah1, false, Bih1[hf + 2].v, (short)0, splat8(bias1[hf + 2]), false, false);
            af     = __builtin_amdgcn_wmma_f32_16x16x32_f16(false, A2,  false, Bhh1[hf + 2].v, (short)0, af, false, false);
            v8f ag = __builtin_amdgcn_wmma_f32_16x16x32_f16(false, Ah1, false, Bih1[hf + 4].v, (short)0, splat8(bias1[hf + 4]), false, false);
            ag     = __builtin_amdgcn_wmma_f32_16x16x32_f16(false, A2,  false, Bhh1[hf + 4].v, (short)0, ag, false, false);
            v8f ao = __builtin_amdgcn_wmma_f32_16x16x32_f16(false, Ah1, false, Bih1[hf + 6].v, (short)0, splat8(bias1[hf + 6]), false, false);
            ao     = __builtin_amdgcn_wmma_f32_16x16x32_f16(false, A2,  false, Bhh1[hf + 6].v, (short)0, ao, false, false);
            V8F vi, vf, vg, vo; vi.v = ai; vf.v = af; vg.v = ag; vo.v = ao;
#pragma unroll
            for (int r = 0; r < 8; ++r) {
                float ig = sigm_f(vi.f[r]);
                float fg = sigm_f(vf.f[r]);
                float gg = tanh_f(vg.f[r]);
                float og = sigm_f(vo.f[r]);
                float cc = fg * cs2[hf].f[r] + ig * gg;
                cs2[hf].f[r] = cc;
                float hh = og * tanh_f(cc);
                h2h[r + 8 * hi][16 * hf + l] = (_Float16)hh;
                if (t == TLEN - 1) h2f[r + 8 * hi][16 * hf + l] = hh;  // uniform branch
            }
        }
        __syncthreads();
    }

    // ---- LayerNorm over H=32 of the final layer-2 hidden state (f32 copy)
    if (lane < 16) {
        float mu = 0.0f;
#pragma unroll
        for (int j = 0; j < HID; ++j) mu += h2f[l][j];
        mu *= (1.0f / HID);
        float var = 0.0f;
#pragma unroll
        for (int j = 0; j < HID; ++j) { float d = h2f[l][j] - mu; var += d * d; }
        var *= (1.0f / HID);
        float inv = rsqrtf(var + 1e-5f);
        float* op = out + (size_t)(b0 + l) * HID;
#pragma unroll
        for (int j = 0; j < HID; ++j)
            op[j] = (h2f[l][j] - mu) * inv * lng[j] + lnb[j];
    }
}

extern "C" void kernel_launch(void* const* d_in, const int* in_sizes, int n_in,
                              void* d_out, int out_size, void* d_ws, size_t ws_size,
                              hipStream_t stream) {
    const float* x    = (const float*)d_in[0];
    const float* Wih0 = (const float*)d_in[1];
    const float* Whh0 = (const float*)d_in[2];
    const float* bih0 = (const float*)d_in[3];
    const float* bhh0 = (const float*)d_in[4];
    const float* Wih1 = (const float*)d_in[5];
    const float* Whh1 = (const float*)d_in[6];
    const float* bih1 = (const float*)d_in[7];
    const float* bhh1 = (const float*)d_in[8];
    const float* lng  = (const float*)d_in[9];
    const float* lnb  = (const float*)d_in[10];
    float* out = (float*)d_out;

    // 4096 batch rows / 16 rows per wave = 256 single-wave workgroups
    lstm2_ln_kernel<<<256, 32, 0, stream>>>(x, Wih0, Whh0, bih0, bhh0,
                                            Wih1, Whh1, bih1, bhh1,
                                            lng, lnb, out);
}
